// TransformerBlock_14980845928562
// MI455X (gfx1250) — matrix operand order repaired and checked by reading, not compiled, not run on hardware
//
#include <hip/hip_runtime.h>
#include <math.h>

#define TOKENS 8192
#define EMB    768
#define HID    3072
#define QKV_N  2304
#define SEQ    1024
#define NHEADS 12
#define HEADD  64
#define BHN    96

typedef __attribute__((ext_vector_type(16))) __bf16       v16bf;
typedef __attribute__((ext_vector_type(8)))  float        v8f;
typedef __attribute__((ext_vector_type(4)))  unsigned int u32x4;

// Exact pointee type of the async-to-LDS builtin params (gcc-style int4).
typedef int v4i_vs __attribute__((vector_size(16)));
typedef __attribute__((address_space(1))) v4i_vs* as1_v4i;
typedef __attribute__((address_space(3))) v4i_vs* as3_v4i;

union FragU  { v16bf v; u32x4 q[2]; unsigned short h[16]; };
union Chunk8 { u32x4 q; unsigned short h[8]; };

#ifndef __has_builtin
#define __has_builtin(x) 0
#endif

#if __has_builtin(__builtin_amdgcn_global_load_async_to_lds_b128)
#define HAVE_ASYNC_LDS 1
#else
#define HAVE_ASYNC_LDS 0
#endif

// 16-byte global -> LDS copy. Prefer the CDNA5 async-to-LDS path (ASYNCcnt,
// no VGPR round trip, overlaps WMMA); fall back to load+ds_store.
__device__ __forceinline__ void async_cp16(const unsigned short* g, unsigned short* l) {
#if HAVE_ASYNC_LDS
  __builtin_amdgcn_global_load_async_to_lds_b128(
      (as1_v4i)(void*)(unsigned short*)g,   // cast away const, then generic->AS1
      (as3_v4i)(void*)l,                    // generic->AS3 (addrspacecast)
      0, 0);
#else
  *(u32x4*)l = *(const u32x4*)g;
#endif
}

__device__ __forceinline__ void wait_async0() {
#if HAVE_ASYNC_LDS
#if __has_builtin(__builtin_amdgcn_s_wait_asynccnt)
  __builtin_amdgcn_s_wait_asynccnt(0);
#else
  asm volatile("s_wait_asynccnt 0" ::: "memory");
#endif
#endif
}

__device__ __forceinline__ unsigned short f2bf(float f) {
  unsigned int u = __float_as_uint(f);
  u += 0x7FFFu + ((u >> 16) & 1u);          // round-to-nearest-even
  return (unsigned short)(u >> 16);
}
__device__ __forceinline__ float bf2f(unsigned short h) {
  return __uint_as_float(((unsigned int)h) << 16);
}
__device__ __forceinline__ float ld_as_f(const float* p, long i)          { return p[i]; }
__device__ __forceinline__ float ld_as_f(const unsigned short* p, long i) { return bf2f(p[i]); }

__device__ __forceinline__ v8f vzero() {
  v8f z = {0.f,0.f,0.f,0.f,0.f,0.f,0.f,0.f};
  return z;
}

__device__ __forceinline__ v8f wmma_bf16(v16bf a, v16bf b, v8f c) {
  // D(f32 16x16) = A(bf16 16x32) * B(bf16 32x16) + C
  return __builtin_amdgcn_wmma_f32_16x16x32_bf16(false, a, false, b, (short)0, c, false, false);
}

// A-operand fragment per ISA 16-bit A 16x32 table:
// lanes 0-15: rows M=0..15, K 0..7 then 16..23; lanes 16-31: same rows, K 8..15 then 24..31.
__device__ __forceinline__ v16bf load_frag_a(const unsigned short* base, int ld) {
  const int lane = threadIdx.x & 31;
  const unsigned short* r = base + (lane & 15) * ld + ((lane >> 4) << 3);
  FragU f;
  f.q[0] = *(const u32x4*)(r);
  f.q[1] = *(const u32x4*)(r + 16);
  return f.v;
}
// B-operand fragment: `base` holds rows of B^T (one row per output column, K contiguous).
// lane n = column; lanes 0-15 K 0..7 then 16..23, lanes 16-31 K 8..15 then 24..31 (B rows striped across lanes).
__device__ __forceinline__ v16bf load_frag_b(const unsigned short* base, int ld) {
  const int lane = threadIdx.x & 31;
  const unsigned short* r = base + (lane & 15) * ld + ((lane >> 4) << 3);
  FragU f;
  f.q[0] = *(const u32x4*)(r);
  f.q[1] = *(const u32x4*)(r + 16);
  return f.v;
}

__device__ __forceinline__ float gelu_f(float x) {
  return 0.5f * x * (1.0f + erff(x * 0.7071067811865475f));   // exact GELU
}

// ---------------------------------------------------------------------------
// Weight prep: W[K][N] f32 -> W^T[N][K] bf16
// ---------------------------------------------------------------------------
__global__ __launch_bounds__(256) void wtrans_kernel(const float* __restrict__ W,
                                                     unsigned short* __restrict__ Wt,
                                                     int K, int N) {
  long idx = (long)blockIdx.x * 256 + threadIdx.x;
  long total = (long)K * N;
  if (idx >= total) return;
  long n = idx / K;
  long k = idx - n * K;
  Wt[idx] = f2bf(W[k * (long)N + n]);
}

// ---------------------------------------------------------------------------
// LayerNorm over last dim D, output bf16.  One block per token, 8 waves.
// ---------------------------------------------------------------------------
template <typename TIN>
__global__ __launch_bounds__(256) void ln_bf16_kernel(const TIN* __restrict__ x,
                                                      const float* __restrict__ gamma,
                                                      const float* __restrict__ beta,
                                                      unsigned short* __restrict__ out,
                                                      int D) {
  __shared__ float redS[8], redS2[8];
  const long t = blockIdx.x;
  const TIN* row = x + t * D;
  float s = 0.f, s2 = 0.f;
  for (int i = threadIdx.x; i < D; i += 256) {
    float v = ld_as_f(row, i);
    s += v; s2 += v * v;
  }
#pragma unroll
  for (int m = 16; m >= 1; m >>= 1) {   // wave32 reduction
    s  += __shfl_xor(s, m, 32);
    s2 += __shfl_xor(s2, m, 32);
  }
  if ((threadIdx.x & 31) == 0) { redS[threadIdx.x >> 5] = s; redS2[threadIdx.x >> 5] = s2; }
  __syncthreads();
  float ts = 0.f, ts2 = 0.f;
#pragma unroll
  for (int i = 0; i < 8; ++i) { ts += redS[i]; ts2 += redS2[i]; }
  const float mean = ts / (float)D;
  const float var  = ts2 / (float)D - mean * mean;
  const float rstd = rsqrtf(var + 1e-5f);
  unsigned short* orow = out + t * D;
  for (int i = threadIdx.x; i < D; i += 256) {
    orow[i] = f2bf((ld_as_f(row, i) - mean) * rstd * gamma[i] + beta[i]);
  }
}

// ---------------------------------------------------------------------------
// WMMA GEMM: C[T][N] = epi(A[T][K]bf16 @ (Bt[N][K])^T + bias)
// 128x128 block tile, 8 waves of 32x64, K-step 32, double-buffered LDS with
// async global->LDS staging overlapping the WMMA loop.
// EPI: 0 = bias->bf16, 1 = bias+GELU->bf16, 2 = bias->f32, 3 = bias->qkv scatter bf16
// ---------------------------------------------------------------------------
template <int EPI>
__global__ __launch_bounds__(256) void gemm_wmma(const unsigned short* __restrict__ A,
                                                 const unsigned short* __restrict__ Bt,
                                                 const float* __restrict__ bias,
                                                 void* __restrict__ Cout,
                                                 int T, int N, int K) {
  __shared__ unsigned short sA[2][128 * 40];   // 32 K + 8 pad (80B rows, 16B aligned)
  __shared__ unsigned short sB[2][128 * 40];
  const int tid  = threadIdx.x;
  const int lane = tid & 31;
  const int wave = tid >> 5;
  const int n0 = blockIdx.x * 128;
  const int m0 = blockIdx.y * 128;
  const int wm = (wave >> 1) * 32;   // 4 row groups
  const int wn = (wave & 1) * 64;    // 2 col groups

  v8f acc[2][4];
#pragma unroll
  for (int i = 0; i < 2; ++i)
#pragma unroll
    for (int j = 0; j < 4; ++j) acc[i][j] = vzero();

  const int KT = K >> 5;

  auto stage = [&](int kt, int buf) {
    const unsigned short* ga = A  + (size_t)m0 * K + ((size_t)kt << 5);
    const unsigned short* gb = Bt + (size_t)n0 * K + ((size_t)kt << 5);
#pragma unroll
    for (int i = tid; i < 512; i += 256) {      // 128 rows x 4 chunks of 8 bf16
      const int r = i >> 2;
      const int c = (i & 3) << 3;
      async_cp16(&ga[(size_t)r * K + c], &sA[buf][r * 40 + c]);
      async_cp16(&gb[(size_t)r * K + c], &sB[buf][r * 40 + c]);
    }
    if (kt + 1 < KT) {                          // hint next tile toward L2
      __builtin_prefetch(&ga[(size_t)(tid >> 1) * K + 32], 0, 1);
      __builtin_prefetch(&gb[(size_t)(tid >> 1) * K + 32], 0, 1);
    }
  };

  stage(0, 0);
  wait_async0();
  __syncthreads();
  for (int kt = 0; kt < KT; ++kt) {
    const int buf = kt & 1;
    if (kt + 1 < KT) stage(kt + 1, buf ^ 1);    // async DMA overlaps WMMAs below
    v16bf a0 = load_frag_a(&sA[buf][wm * 40], 40);
    v16bf a1 = load_frag_a(&sA[buf][(wm + 16) * 40], 40);
#pragma unroll
    for (int j = 0; j < 4; ++j) {
      v16bf bfr = load_frag_b(&sB[buf][(wn + j * 16) * 40], 40);
      acc[0][j] = wmma_bf16(a0, bfr, acc[0][j]);
      acc[1][j] = wmma_bf16(a1, bfr, acc[1][j]);
    }
    wait_async0();                              // next buffer landed in LDS
    __syncthreads();
  }

  // epilogue (C layout: VGPR r -> lanes 0-15: M=r,N=lane ; lanes 16-31: M=8+r,N=lane-16)
  const int rr = (lane >> 4) << 3;
  const int cc = lane & 15;
#pragma unroll
  for (int i = 0; i < 2; ++i) {
#pragma unroll
    for (int j = 0; j < 4; ++j) {
      const int col = n0 + wn + j * 16 + cc;
      const float bv = bias[col];
#pragma unroll
      for (int r = 0; r < 8; ++r) {
        const int row = m0 + wm + i * 16 + rr + r;
        float v = acc[i][j][r] + bv;
        if (EPI == 1) v = gelu_f(v);
        if (EPI == 2) {
          ((float*)Cout)[(size_t)row * N + col] = v;
        } else if (EPI == 3) {
          const int which = col / EMB;            // 0=q 1=k 2=v
          const int rem   = col - which * EMB;
          const int head  = rem >> 6;
          const int s     = rem & 63;
          const int bb    = row >> 10;            // token -> (batch, l)
          const int l     = row & 1023;
          ((unsigned short*)Cout)[(size_t)which * ((size_t)BHN * SEQ * HEADD) +
                                  ((size_t)(bb * NHEADS + head) * SEQ + l) * HEADD + s] = f2bf(v);
        } else {
          ((unsigned short*)Cout)[(size_t)row * N + col] = f2bf(v);
        }
      }
    }
  }
}

// ---------------------------------------------------------------------------
// Attention pass A: per-row (t) online softmax stats  M[t], Z[t]
// score[t,l] = sum_c q[t,c]k[l,c];  softmax over l  (reference axis=2)
// grid (96, SEQ/128); wave w owns t-rows [t0+16w, t0+16w+16)
// ---------------------------------------------------------------------------
__global__ __launch_bounds__(256) void attn_stats_kernel(const unsigned short* __restrict__ q,
                                                         const unsigned short* __restrict__ kmat,
                                                         float* __restrict__ Mrow,
                                                         float* __restrict__ Zrow,
                                                         float scale) {
  __shared__ unsigned short sK[128 * 72];
  const int bh   = blockIdx.x;
  const int t0   = blockIdx.y * 128;
  const int tid  = threadIdx.x;
  const int lane = tid & 31;
  const int wave = tid >> 5;
  const unsigned short* qb = q    + (size_t)bh * SEQ * HEADD;
  const unsigned short* kb = kmat + (size_t)bh * SEQ * HEADD;

  v16bf qa0 = load_frag_a(qb + (size_t)(t0 + wave * 16) * HEADD, HEADD);        // k 0..31
  v16bf qa1 = load_frag_a(qb + (size_t)(t0 + wave * 16) * HEADD + 32, HEADD);   // k 32..63

  float rm[8], rs[8];
#pragma unroll
  for (int r = 0; r < 8; ++r) { rm[r] = -1e30f; rs[r] = 0.f; }

  for (int l0 = 0; l0 < SEQ; l0 += 128) {
    __syncthreads();
    for (int i = tid; i < 1024; i += 256) {      // stage K tile 128x64 bf16 (async)
      const int r = i >> 3;
      const int c = (i & 7) << 3;
      async_cp16(&kb[(size_t)(l0 + r) * HEADD + c], &sK[r * 72 + c]);
    }
    wait_async0();
    __syncthreads();
#pragma unroll
    for (int j = 0; j < 8; ++j) {
      v16bf b0 = load_frag_b(&sK[(j * 16) * 72], 72);
      v16bf b1 = load_frag_b(&sK[(j * 16) * 72 + 32], 72);
      v8f c = vzero();
      c = wmma_bf16(qa0, b0, c);
      c = wmma_bf16(qa1, b1, c);
#pragma unroll
      for (int r = 0; r < 8; ++r) {              // online max/sum over the 16 cols (half-wave)
        float v = c[r] * scale;
        float tm = v;
#pragma unroll
        for (int m = 8; m >= 1; m >>= 1) tm = fmaxf(tm, __shfl_xor(tm, m, 32));
        const float mn = fmaxf(rm[r], tm);
        float e = __expf(v - mn);
#pragma unroll
        for (int m = 8; m >= 1; m >>= 1) e += __shfl_xor(e, m, 32);
        rs[r] = rs[r] * __expf(rm[r] - mn) + e;
        rm[r] = mn;
      }
    }
  }
  if ((lane & 15) == 0) {
    const int rbase = t0 + wave * 16 + ((lane >> 4) << 3);
#pragma unroll
    for (int r = 0; r < 8; ++r) {
      Mrow[(size_t)bh * SEQ + rbase + r] = rm[r];
      Zrow[(size_t)bh * SEQ + rbase + r] = rs[r];
    }
  }
}

// ---------------------------------------------------------------------------
// Attention pass B: O[l,c] = sum_t P[t,l] v[t,c]  (= P^T V), P = exp(s-m)/Z
// grid (96, SEQ/128): block owns 128 l-rows; loops t-tiles of 128.
// P^T and V^T staged through LDS for the transposed contraction.
// Output scattered to token-major [B*SEQ][768] bf16 (feature = head*64+c).
// ---------------------------------------------------------------------------
__global__ __launch_bounds__(256) void attn_out_kernel(const unsigned short* __restrict__ q,
                                                       const unsigned short* __restrict__ kmat,
                                                       const unsigned short* __restrict__ vmat,
                                                       const float* __restrict__ Mrow,
                                                       const float* __restrict__ Zrow,
                                                       unsigned short* __restrict__ out,
                                                       float scale) {
  __shared__ unsigned short sP[128 * 136];   // P^T: [l_local][t_local]
  __shared__ unsigned short sVt[64 * 136];   // V^T: [c][t_local]
  const int bh   = blockIdx.x;
  const int b    = bh / NHEADS;
  const int head = bh - b * NHEADS;
  const int l0   = blockIdx.y * 128;
  const int tid  = threadIdx.x;
  const int lane = tid & 31;
  const int wave = tid >> 5;
  const int rr = (lane >> 4) << 3;
  const int cc = lane & 15;
  const unsigned short* qb = q    + (size_t)bh * SEQ * HEADD;
  const unsigned short* kb = kmat + (size_t)bh * SEQ * HEADD;
  const unsigned short* vb = vmat + (size_t)bh * SEQ * HEADD;

  v8f oacc[4];
#pragma unroll
  for (int j = 0; j < 4; ++j) oacc[j] = vzero();

  for (int t0 = 0; t0 < SEQ; t0 += 128) {
    __syncthreads();
    // stage V^T tile (transposing scatter -> keep as ds stores)
    for (int i = tid; i < 1024; i += 256) {
      const int t  = i >> 3;
      const int c0 = (i & 7) << 3;
      Chunk8 ch;
      ch.q = *(const u32x4*)&vb[(size_t)(t0 + t) * HEADD + c0];
#pragma unroll
      for (int e = 0; e < 8; ++e) sVt[(c0 + e) * 136 + t] = ch.h[e];
    }
    // S tiles for this wave's 16 t-rows, all 128 l; write P^T into LDS
    v16bf a0 = load_frag_a(qb + (size_t)(t0 + wave * 16) * HEADD, HEADD);
    v16bf a1 = load_frag_a(qb + (size_t)(t0 + wave * 16) * HEADD + 32, HEADD);
    float mr[8], zr[8];
    {
      const int rbase = t0 + wave * 16 + rr;
#pragma unroll
      for (int r = 0; r < 8; ++r) {
        mr[r] = Mrow[(size_t)bh * SEQ + rbase + r];
        zr[r] = Zrow[(size_t)bh * SEQ + rbase + r];
      }
    }
#pragma unroll
    for (int j = 0; j < 8; ++j) {
      v16bf b0 = load_frag_b(kb + (size_t)(l0 + j * 16) * HEADD, HEADD);
      v16bf b1 = load_frag_b(kb + (size_t)(l0 + j * 16) * HEADD + 32, HEADD);
      v8f c = vzero();
      c = wmma_bf16(a0, b0, c);
      c = wmma_bf16(a1, b1, c);
#pragma unroll
      for (int r = 0; r < 8; ++r) {
        const float p = __expf(c[r] * scale - mr[r]) / zr[r];
        sP[(j * 16 + cc) * 136 + wave * 16 + rr + r] = f2bf(p);   // transposed store
      }
    }
    __syncthreads();
    // O += P^T V  : A rows = l (from sP), B = V (columns from sVt), 4 K-steps of 32 t
#pragma unroll
    for (int kt = 0; kt < 4; ++kt) {
      v16bf pa = load_frag_a(&sP[(wave * 16) * 136 + kt * 32], 136);
#pragma unroll
      for (int j = 0; j < 4; ++j) {
        v16bf vf = load_frag_b(&sVt[(j * 16) * 136 + kt * 32], 136);
        oacc[j] = wmma_bf16(pa, vf, oacc[j]);
      }
    }
  }

#pragma unroll
  for (int j = 0; j < 4; ++j)
#pragma unroll
    for (int r = 0; r < 8; ++r) {
      const int l = l0 + wave * 16 + rr + r;
      out[(size_t)(b * SEQ + l) * EMB + head * HEADD + j * 16 + cc] = f2bf(oacc[j][r]);
    }
}

// ---------------------------------------------------------------------------
// Host-side orchestration
// ---------------------------------------------------------------------------
extern "C" void kernel_launch(void* const* d_in, const int* in_sizes, int n_in,
                              void* d_out, int out_size, void* d_ws, size_t ws_size,
                              hipStream_t stream) {
  (void)in_sizes; (void)n_in; (void)out_size; (void)ws_size;

  const float* x       = (const float*)d_in[0];
  const float* qkv_g   = (const float*)d_in[1];
  const float* qkv_be  = (const float*)d_in[2];
  const float* qkv_W1  = (const float*)d_in[3];
  const float* qkv_b1  = (const float*)d_in[4];
  const float* qkv_W2  = (const float*)d_in[5];
  const float* qkv_b2  = (const float*)d_in[6];
  const float* proj_g  = (const float*)d_in[7];
  const float* proj_be = (const float*)d_in[8];
  const float* proj_W1 = (const float*)d_in[9];
  const float* proj_b1 = (const float*)d_in[10];
  const float* proj_W2 = (const float*)d_in[11];
  const float* proj_b2 = (const float*)d_in[12];
  const float* mlp_g   = (const float*)d_in[13];
  const float* mlp_be  = (const float*)d_in[14];
  const float* mlp_W1  = (const float*)d_in[15];
  const float* mlp_b1  = (const float*)d_in[16];
  const float* mlp_W2  = (const float*)d_in[17];
  const float* mlp_b2  = (const float*)d_in[18];

  size_t off = 0;
  auto walloc = [&](size_t bytes) -> void* {
    void* p = (char*)d_ws + off;
    off += (bytes + 255) & ~(size_t)255;
    return p;
  };

  unsigned short* qkvW1t  = (unsigned short*)walloc((size_t)EMB * HID   * 2);
  unsigned short* qkvW2t  = (unsigned short*)walloc((size_t)HID * QKV_N * 2);
  unsigned short* projW1t = (unsigned short*)walloc((size_t)EMB * HID   * 2);
  unsigned short* projW2t = (unsigned short*)walloc((size_t)HID * EMB   * 2);
  unsigned short* mlpW1t  = (unsigned short*)walloc((size_t)EMB * HID   * 2);
  unsigned short* mlpW2t  = (unsigned short*)walloc((size_t)HID * EMB   * 2);
  unsigned short* bufLN   = (unsigned short*)walloc((size_t)TOKENS * EMB * 2);
  unsigned short* bufHid  = (unsigned short*)walloc((size_t)TOKENS * HID * 2);
  unsigned short* bufQKV  = (unsigned short*)walloc((size_t)3 * BHN * SEQ * HEADD * 2);
  float*          Mrow    = (float*)walloc((size_t)BHN * SEQ * 4);
  float*          Zrow    = (float*)walloc((size_t)BHN * SEQ * 4);
  unsigned short* bufTok  = (unsigned short*)walloc((size_t)TOKENS * EMB * 2);

  auto launch_wt = [&](const float* W, unsigned short* Wt, int K, int N) {
    long total = (long)K * N;
    wtrans_kernel<<<dim3((unsigned)((total + 255) / 256)), dim3(256), 0, stream>>>(W, Wt, K, N);
  };
  launch_wt(qkv_W1,  qkvW1t,  EMB, HID);
  launch_wt(qkv_W2,  qkvW2t,  HID, QKV_N);
  launch_wt(proj_W1, projW1t, EMB, HID);
  launch_wt(proj_W2, projW2t, HID, EMB);
  launch_wt(mlp_W1,  mlpW1t,  EMB, HID);
  launch_wt(mlp_W2,  mlpW2t,  HID, EMB);

  const float scale = 0.125f;  // 1/sqrt(64)

  // ---- block 1: qkv ----
  ln_bf16_kernel<float><<<dim3(TOKENS), dim3(256), 0, stream>>>(x, qkv_g, qkv_be, bufLN, EMB);
  gemm_wmma<1><<<dim3(HID / 128, TOKENS / 128), dim3(256), 0, stream>>>(
      bufLN, qkvW1t, qkv_b1, (void*)bufHid, TOKENS, HID, EMB);
  gemm_wmma<3><<<dim3(QKV_N / 128, TOKENS / 128), dim3(256), 0, stream>>>(
      bufHid, qkvW2t, qkv_b2, (void*)bufQKV, TOKENS, QKV_N, HID);

  // ---- attention ----
  const unsigned short* qp = bufQKV;
  const unsigned short* kp = bufQKV + (size_t)BHN * SEQ * HEADD;
  const unsigned short* vp = bufQKV + (size_t)2 * BHN * SEQ * HEADD;
  attn_stats_kernel<<<dim3(BHN, SEQ / 128), dim3(256), 0, stream>>>(qp, kp, Mrow, Zrow, scale);
  attn_out_kernel<<<dim3(BHN, SEQ / 128), dim3(256), 0, stream>>>(qp, kp, vp, Mrow, Zrow, bufTok, scale);

  // ---- block 2: proj ----
  ln_bf16_kernel<unsigned short><<<dim3(TOKENS), dim3(256), 0, stream>>>(bufTok, proj_g, proj_be, bufLN, EMB);
  gemm_wmma<1><<<dim3(HID / 128, TOKENS / 128), dim3(256), 0, stream>>>(
      bufLN, projW1t, proj_b1, (void*)bufHid, TOKENS, HID, EMB);
  unsigned short* bufProj = bufQKV;  // qkv dead after attention; reuse
  gemm_wmma<0><<<dim3(EMB / 128, TOKENS / 128), dim3(256), 0, stream>>>(
      bufHid, projW2t, proj_b2, (void*)bufProj, TOKENS, EMB, HID);

  // ---- block 3: mlp ----
  ln_bf16_kernel<unsigned short><<<dim3(TOKENS), dim3(256), 0, stream>>>(bufProj, mlp_g, mlp_be, bufLN, EMB);
  gemm_wmma<1><<<dim3(HID / 128, TOKENS / 128), dim3(256), 0, stream>>>(
      bufLN, mlpW1t, mlp_b1, (void*)bufHid, TOKENS, HID, EMB);
  gemm_wmma<2><<<dim3(EMB / 128, TOKENS / 128), dim3(256), 0, stream>>>(
      bufHid, mlpW2t, mlp_b2, d_out, TOKENS, EMB, HID);
}
